// EiConvLayer_18957985645021
// MI455X (gfx1250) — matrix operand order repaired and checked by reading, not compiled, not run on hardware
//
#include <hip/hip_runtime.h>
#include <hip/hip_bf16.h>

// ---------------------------------------------------------------------------
// EI conv layer, fused:
//   Wz = We - Wei @ Wi            (per output channel e, flattened K = Cin*3*3)
//   Wg = (Wei * exp(alpha)) @ Wi
//   z  = conv(x, Wz) ; gamma = conv(x, Wg) + EPS
//   out = relu(g * z / gamma + b)
// One implicit GEMM, bf16 WMMA (v_wmma_f32_16x16x32_bf16), fp32 accumulate,
// double-buffered LDS patch tile with register staging.
// Block: 512 threads (16 waves) -> 128-pixel x 32-channel tile (x2 matrices).
// ---------------------------------------------------------------------------

typedef __attribute__((ext_vector_type(16))) __bf16 v16bf;
typedef __attribute__((ext_vector_type(8)))  float  v8f;

#define EPS_   1e-8f
#define B_     16
#define CIN_   64
#define HIN_   112
#define WIN_   112
#define E_     128
#define I_     16
#define OH_    110
#define OW_    110
#define KK_    576              // CIN_*3*3
#define NSTEP_ (KK_/32)         // 18
#define PIX_   (B_*OH_*OW_)     // 193600
#define PXT_   128              // pixel tile per block (8 sub-tiles x 16)
#define CHT_   32               // channel tile per block (2 sub-tiles x 16)
#define KSTEP_ 32
#define CHW_   (HIN_*WIN_)      // 12544

// ---------------- weight-prep kernel: fuse + convert to bf16 ----------------
__global__ void ei_prep_weights(const float* __restrict__ We,
                                const float* __restrict__ Wi,
                                const float* __restrict__ Wei,
                                const float* __restrict__ alpha,
                                __hip_bfloat16* __restrict__ wz,
                                __hip_bfloat16* __restrict__ wg) {
    int idx = blockIdx.x * blockDim.x + threadIdx.x;   // e*KK_ + k
    if (idx >= E_ * KK_) return;
    int e = idx / KK_;
    int k = idx - e * KK_;
    float sz = We[idx];            // We is [E][Cin][3][3] == [E][KK_]
    float ss = 0.f, sg = 0.f;
#pragma unroll
    for (int i = 0; i < I_; ++i) {
        float w = Wi[i * KK_ + k];
        float c = Wei[e * I_ + i];
        ss += c * w;
        sg += c * __expf(alpha[i]) * w;
    }
    wz[idx] = __float2bfloat16(sz - ss);
    wg[idx] = __float2bfloat16(sg);
}

// ------------------------- main implicit-GEMM kernel ------------------------
// grid.x : pixel tiles (ceil(PIX_/128)), grid.y : 4 channel tiles of 32.
// Wave C tile = 16 channels (M, VGPR axis) x 16 pixels (N, lane axis); two
// accumulators (z and gamma) share the patch (B) fragment.
__global__ __launch_bounds__(512)
void ei_conv_wmma(const float* __restrict__ x,
                  const __hip_bfloat16* __restrict__ wz,
                  const __hip_bfloat16* __restrict__ wg,
                  const float* __restrict__ gvec,
                  const float* __restrict__ bvec,
                  float* __restrict__ out) {
    __shared__ __align__(16) __hip_bfloat16 lW[2 * CHT_][KK_ + 8];      // z rows 0..31, g rows 32..63
    __shared__ __align__(16) __hip_bfloat16 lA[2][PXT_][KSTEP_ + 8];    // double-buffered patch tile

    const int tid    = threadIdx.x;
    const int lane   = tid & 31;            // wave32
    const int waveId = tid >> 5;            // 0..15
    const int pxSub  = waveId & 7;          // 8 x 16-pixel sub-tiles
    const int chSub  = waveId >> 3;         // 2 x 16-channel sub-tiles
    const int half   = lane >> 4;
    const int nIdx   = lane & 15;
    const int chBase = blockIdx.y * CHT_;

    // ---- stage fused weight panel for this channel tile (once) ----
    for (int t = tid; t < 2 * CHT_ * (KK_ / 4); t += 512) {
        int row = t / (KK_ / 4);
        int c4  = (t - row * (KK_ / 4)) * 4;
        const __hip_bfloat16* src = (row < CHT_)
            ? (wz + (size_t)(chBase + row) * KK_ + c4)
            : (wg + (size_t)(chBase + row - CHT_) * KK_ + c4);
        *reinterpret_cast<unsigned long long*>(&lW[row][c4]) =
            *reinterpret_cast<const unsigned long long*>(src);
    }

    // ---- patch loader setup: per-lane pixel base + wave-uniform k-walk ----
    // Lanes cover 32 consecutive pixels (coalesced 128B loads); the 8-wide
    // k-chunk (0/8/16/24) is uniform per wave -> scalar wrap logic.
    const int pxl  = tid & 127;
    const int ksub = ((tid >> 7) & 3) * 8;  // 0/8/16/24, wave-uniform
    int pg = blockIdx.x * PXT_ + pxl;
    pg = (pg < PIX_) ? pg : (PIX_ - 1);     // clamp, keep EXEC full
    int bb  = pg / (OH_ * OW_);
    int rem = pg - bb * (OH_ * OW_);
    int ohp = rem / OW_;
    int owp = rem - ohp * OW_;
    const float* xpx = x + (size_t)bb * CIN_ * CHW_ + (size_t)ohp * WIN_ + owp;

    // issue 8 global loads for K range [k0+ksub, k0+ksub+8) into regs
    auto issueLoad = [&](int k0, float* r) {
        int k   = k0 + ksub;
        int cin = k / 9;
        int rr  = k - cin * 9;
        int kh  = rr / 3;
        int kw  = rr - kh * 3;
        int off = cin * CHW_ + kh * WIN_ + kw;
#pragma unroll
        for (int j = 0; j < 8; ++j) {
            r[j] = xpx[off];
            // incremental im2col walk: k -> k+1
            off += 1;
            kw += 1;
            if (kw == 3) {                   // kw wrap: next image row
                kw = 0; kh += 1; off += WIN_ - 3;          // +109
                if (kh == 3) {               // kh wrap: next input channel
                    kh = 0; off += CHW_ - 3 * WIN_;        // +12208
                }
            }
        }
    };
    // convert+pack regs to bf16 pairs, store to LDS buffer
    auto storeLds = [&](int buf, const float* r) {
#pragma unroll
        for (int j = 0; j < 8; j += 2) {
            __hip_bfloat162 p = __float22bfloat162_rn(float2{r[j], r[j + 1]});
            *reinterpret_cast<unsigned*>(&lA[buf][pxl][ksub + j]) =
                *reinterpret_cast<unsigned*>(&p);
        }
    };

    v8f cz = {};
    v8f cg = {};

    // fragment gather + 2 WMMAs for one 32-wide K step
    auto compute = [&](int buf, int k0) {
        union { v16bf v; unsigned u[8]; } az, ag, bp;
#pragma unroll
        for (int j = 0; j < 8; ++j) {
            // A (16x32, M=channel): pair-packed K, lane-halves offset by 8
            int kbA = ((j < 4) ? (2 * j) : (16 + 2 * (j - 4))) + (half ? 8 : 0);
            az.u[j] = *reinterpret_cast<const unsigned*>(
                          &lW[chSub * 16 + nIdx][k0 + kbA]);
            ag.u[j] = *reinterpret_cast<const unsigned*>(
                          &lW[CHT_ + chSub * 16 + nIdx][k0 + kbA]);
            // B (32x16, N=pixel): pair-packed K, lane-halves offset by 8 (same as A)
            int kbB = kbA;
            bp.u[j] = *reinterpret_cast<const unsigned*>(
                          &lA[buf][pxSub * 16 + nIdx][kbB]);
        }
        cz = __builtin_amdgcn_wmma_f32_16x16x32_bf16(
                false, az.v, false, bp.v, (short)0, cz, false, false);
        cg = __builtin_amdgcn_wmma_f32_16x16x32_bf16(
                false, ag.v, false, bp.v, (short)0, cg, false, false);
    };

    // ---- software-pipelined main loop: 18 K-steps, x2 unrolled, 2 buffers ----
    float r0[8], r1[8];
    issueLoad(0, r0);
#pragma unroll 1
    for (int p = 0; p < NSTEP_ / 2; ++p) {
        int sA = 2 * p, sB = 2 * p + 1;
        // step A: buffer 0 / regs r0; prefetch next step into r1
        storeLds(0, r0);
        issueLoad((sA + 1) * KSTEP_, r1);        // (2p+1)*32 <= 544 < 576, always valid
        __syncthreads();
        compute(0, sA * KSTEP_);
        // step B: buffer 1 / regs r1; prefetch next step into r0
        storeLds(1, r1);
        if (sB + 1 < NSTEP_) issueLoad((sB + 1) * KSTEP_, r0);
        __syncthreads();
        compute(1, sB * KSTEP_);
    }

    // ---- epilogue: out = relu(g * z / (gamma + EPS) + b) ----
    int pge  = blockIdx.x * PXT_ + pxSub * 16 + nIdx;    // C lane axis = pixel
    bool ok  = pge < PIX_;
    int pgc  = ok ? pge : (PIX_ - 1);
    int bo   = pgc / (OH_ * OW_);
    int re   = pgc - bo * (OH_ * OW_);
    int oh   = re / OW_;
    int ow   = re - oh * OW_;
#pragma unroll
    for (int v = 0; v < 8; ++v) {
        int ch = chBase + chSub * 16 + v + (half ? 8 : 0);   // C VGPR axis = channel
        float val = gvec[ch] * (cz[v] / (cg[v] + EPS_)) + bvec[ch];
        val = fmaxf(val, 0.f);
        if (ok)
            out[(((size_t)bo * E_ + ch) * OH_ + oh) * OW_ + ow] = val;
    }
}

// ------------------------------- launcher -----------------------------------
extern "C" void kernel_launch(void* const* d_in, const int* in_sizes, int n_in,
                              void* d_out, int out_size, void* d_ws, size_t ws_size,
                              hipStream_t stream) {
    const float* x     = (const float*)d_in[0];
    const float* We    = (const float*)d_in[1];
    const float* Wi    = (const float*)d_in[2];
    const float* Wei   = (const float*)d_in[3];
    const float* alpha = (const float*)d_in[4];
    const float* g     = (const float*)d_in[5];
    const float* b     = (const float*)d_in[6];
    float* out = (float*)d_out;

    // workspace: fused bf16 weights, Wz then Wg, each [E_][KK_]  (~288 KB)
    __hip_bfloat16* wz = (__hip_bfloat16*)d_ws;
    __hip_bfloat16* wg = wz + (size_t)E_ * KK_;

    {
        int total = E_ * KK_;
        dim3 grid((total + 255) / 256);
        ei_prep_weights<<<grid, 256, 0, stream>>>(We, Wi, Wei, alpha, wz, wg);
    }
    {
        dim3 grid((PIX_ + PXT_ - 1) / PXT_, E_ / CHT_, 1);   // 1513 x 4
        ei_conv_wmma<<<grid, 512, 0, stream>>>(x, wz, wg, g, b, out);
    }
}
